// AudNet_82300163326026
// MI455X (gfx1250) — matrix operand order repaired and checked by reading, not compiled, not run on hardware
//
#include <hip/hip_runtime.h>
#include <hip/hip_bf16.h>

typedef __bf16 bf16;
typedef __attribute__((ext_vector_type(2)))  float v2f;
typedef __attribute__((ext_vector_type(4)))  float v4f;
typedef __attribute__((ext_vector_type(8)))  float v8f;
typedef __attribute__((ext_vector_type(16))) bf16  v16bf;
typedef __attribute__((ext_vector_type(8)))  bf16  v8bf;

// ---------------- dimensions ----------------
#define NSTEPS 81
#define NFREQ  129
#define H      1000
#define HP     1008        // 63 N-tiles of 16
#define NTILES (HP / 16)   // 63
#define KBF    1024        // K padded for bf16 WMMA (32 k-steps of 32)
#define KFP    132         // layer-1 K padded to 33 k-steps of 4
#define HL     20
#define HLP    32
#define OUTN   10
#define OUTP   16
#define NROW   16          // batch rows per workgroup

// ---------------- workspace layout (byte offsets) ----------------
#define WSB_W1T  0                            // f32 [132][1008] (k-major, layer-1 B frags)
#define WSB_B1   (WSB_W1T + KFP*HP*4)         // f32 [1008]
#define WSB_B2   (WSB_B1 + HP*4)              // f32 [1008]
#define WSB_B3   (WSB_B2 + HP*4)              // f32 [32]
#define WSB_B4   (WSB_B3 + HLP*4)             // f32 [16]
#define WSB_W4T  (WSB_B4 + OUTP*4)            // f32 [20][16] (k-major)
#define WSB_W2HI (WSB_W4T + HL*OUTP*4)        // bf16 [1008][1024] row-major (32B aligned)
#define WSB_W2LO (WSB_W2HI + HP*KBF*2)        // bf16 [1008][1024]
#define WSB_W3HI (WSB_W2LO + HP*KBF*2)        // bf16 [32][1024]
#define WSB_W3LO (WSB_W3HI + HLP*KBF*2)       // bf16 [32][1024]
#define WS_BYTES (WSB_W3LO + HLP*KBF*2)

#define PREP_MAX (HP*KBF)                     // largest region element count (W2)

// ---------------- weight transpose / bf16-split / pad kernel ----------------
__global__ void snn_prep(const float* __restrict__ W1, const float* __restrict__ b1,
                         const float* __restrict__ W2, const float* __restrict__ b2,
                         const float* __restrict__ W3, const float* __restrict__ b3,
                         const float* __restrict__ W4, const float* __restrict__ b4,
                         char* __restrict__ wsb) {
    long i = (long)blockIdx.x * blockDim.x + threadIdx.x;
    float* W1T = (float*)(wsb + WSB_W1T);
    float* b1p = (float*)(wsb + WSB_B1);
    float* b2p = (float*)(wsb + WSB_B2);
    float* b3p = (float*)(wsb + WSB_B3);
    float* b4p = (float*)(wsb + WSB_B4);
    float* W4T = (float*)(wsb + WSB_W4T);
    bf16* W2hi = (bf16*)(wsb + WSB_W2HI);
    bf16* W2lo = (bf16*)(wsb + WSB_W2LO);
    bf16* W3hi = (bf16*)(wsb + WSB_W3HI);
    bf16* W3lo = (bf16*)(wsb + WSB_W3LO);

    if (i < KFP * HP) {                       // W1T [132][1008], zero-padded
        int k = (int)(i / HP), n = (int)(i % HP);
        W1T[i] = (k < NFREQ && n < H) ? W1[(long)n * NFREQ + k] : 0.0f;
    }
    if (i < HP)  { b1p[i] = (i < H)    ? b1[i] : 0.0f;
                   b2p[i] = (i < H)    ? b2[i] : 0.0f; }
    if (i < HLP) { b3p[i] = (i < HL)   ? b3[i] : 0.0f; }
    if (i < OUTP){ b4p[i] = (i < OUTN) ? b4[i] : 0.0f; }
    if (i < HL * OUTP) {                      // W4T [20][16] k-major
        int k = (int)(i / OUTP), n = (int)(i % OUTP);
        W4T[i] = (n < OUTN) ? W4[(long)n * HL + k] : 0.0f;
    }
    if (i < (long)HP * KBF) {                 // W2 hi/lo split, row-major, padded
        int n = (int)(i / KBF), k = (int)(i % KBF);
        float w = (n < H && k < H) ? W2[(long)n * H + k] : 0.0f;
        bf16 hi = (bf16)w;
        W2hi[i] = hi;
        W2lo[i] = (bf16)(w - (float)hi);
    }
    if (i < (long)HLP * KBF) {                // W3 hi/lo split
        int n = (int)(i / KBF), k = (int)(i % KBF);
        float w = (n < HL && k < H) ? W3[(long)n * H + k] : 0.0f;
        bf16 hi = (bf16)w;
        W3hi[i] = hi;
        W3lo[i] = (bf16)(w - (float)hi);
    }
}

// ---------------- persistent SNN kernel ----------------
// grid = n/16 workgroups, 256 threads (8 wave32). Each WG owns 16 batch rows
// and runs the entire 81-step recurrence with state resident in LDS/registers.
// Layer 1 & 4: exact f32 WMMA (16x16x4). Layers 2 & 3: split-bf16 WMMA
// (16x16x32), exact for {0,1} spike activations, ~16-bit-mantissa weights.
// Membrane buffers are TILE-MAJOR [(tile*16+n)*16+row] so each lane's 8 state
// values are contiguous 32B -> ds_load/store_b128 instead of 8 scalar b32.
__global__ __launch_bounds__(256, 1)
void snn_main(const float* __restrict__ x, const char* __restrict__ wsb,
              float* __restrict__ out) {
    extern __shared__ float sm[];
    float* mem1 = sm;                                   // f32 63*16*16 (tile-major)
    float* mem2 = sm + NTILES * 256;                    // f32 63*16*16
    bf16*  sAbf = (bf16*)(sm + 2 * NTILES * 256);       // bf16 16*1024 [row][k]
    bf16*  sBbf = sAbf + NROW * KBF;                    // bf16 16*1024 [row][k]
    float* xbuf = (float*)(sBbf + NROW * KBF);          // f32 16*132   [row][k]
    float* s3   = xbuf + NROW * KFP;                    // f32 16*32    [row][n]

    const float* W1T = (const float*)(wsb + WSB_W1T);
    const float* b1p = (const float*)(wsb + WSB_B1);
    const float* b2p = (const float*)(wsb + WSB_B2);
    const float* b3p = (const float*)(wsb + WSB_B3);
    const float* b4p = (const float*)(wsb + WSB_B4);
    const float* W4T = (const float*)(wsb + WSB_W4T);
    const bf16* W2hi = (const bf16*)(wsb + WSB_W2HI);
    const bf16* W2lo = (const bf16*)(wsb + WSB_W2LO);
    const bf16* W3hi = (const bf16*)(wsb + WSB_W3HI);
    const bf16* W3lo = (const bf16*)(wsb + WSB_W3LO);

    const int tid  = threadIdx.x;
    const int lane = tid & 31;
    const int wv   = tid >> 5;        // 8 waves
    const int half = lane >> 4;       // lane-half
    const int l15  = lane & 15;
    const int r0   = blockIdx.x * NROW;

    // zero membrane state and spike buffers (incl. K padding region)
    for (int i = tid; i < NTILES * 256; i += 256) { mem1[i] = 0.0f; mem2[i] = 0.0f; }
    for (int i = tid; i < NROW * KBF; i += 256) { sAbf[i] = (bf16)0.0f; sBbf[i] = (bf16)0.0f; }
    v8f m3 = {0.f,0.f,0.f,0.f,0.f,0.f,0.f,0.f};   // layer-3 state (waves 0,1)
    v8f m4 = {0.f,0.f,0.f,0.f,0.f,0.f,0.f,0.f};   // layer-4 state (wave 0)
    __syncthreads();

    const float BETA = 0.95f;

    for (int t = 0; t < NSTEPS; ++t) {
        // stage x_t tile: x[row, f, t] = x[row*10449 + f*81 + t]
        for (int i = tid; i < NROW * KFP; i += 256) {
            int r = i / KFP, f = i % KFP;
            xbuf[i] = (f < NFREQ)
                ? x[(long)(r0 + r) * (NSTEPS * NFREQ) + (long)f * NSTEPS + t]
                : 0.0f;
        }
        __syncthreads();

        // ---- Layer 1 (f32 WMMA): cur = x_t @ W1^T + b1 ; LIF -> sAbf, mem1 ----
        for (int nt = wv; nt < NTILES; nt += 8) {
            const int n = nt * 16 + l15;
            const float bv = b1p[n];
            v8f c = {bv,bv,bv,bv,bv,bv,bv,bv};
            const float* arow = &xbuf[l15 * KFP];
            for (int k = 0; k < KFP; k += 4) {
                const int kk = k + half * 2;
                v2f a = *(const v2f*)(arow + kk);
                v2f b; b.x = W1T[(long)kk * HP + n]; b.y = W1T[(long)(kk + 1) * HP + n];
                c = __builtin_amdgcn_wmma_f32_16x16x4_f32(false, a, false, b,
                                                          (short)0, c, false, false);
            }
            // LIF: contiguous 32B membrane slice per lane (tile-major layout)
            float* mrow = &mem1[(nt * 16 + l15) * 16 + half * 8];
            v4f m0 = *(v4f*)(mrow);
            v4f m1 = *(v4f*)(mrow + 4);
            #pragma unroll
            for (int v = 0; v < 8; ++v) {
                float m  = (v < 4) ? m0[v & 3] : m1[v & 3];
                float mn = BETA * m + c[v] - ((m > 1.0f) ? 1.0f : 0.0f);
                if (v < 4) m0[v & 3] = mn; else m1[v & 3] = mn;
                sAbf[(v + half * 8) * KBF + n] = (bf16)((mn > 1.0f) ? 1.0f : 0.0f);
            }
            *(v4f*)(mrow) = m0;
            *(v4f*)(mrow + 4) = m1;
        }
        __syncthreads();

        // ---- Layer 2 (split-bf16 WMMA): cur = s1 @ W2^T + b2 ; LIF -> sBbf ----
        for (int nt = wv; nt < NTILES; nt += 8) {
            const int n = nt * 16 + l15;
            if (nt + 8 < NTILES) {    // prefetch next tile's weight rows into caches
                __builtin_prefetch((const void*)(W2hi + (long)(n + 128) * KBF), 0, 0);
                __builtin_prefetch((const void*)(W2lo + (long)(n + 128) * KBF), 0, 0);
            }
            const float bv = b2p[n];
            v8f c = {bv,bv,bv,bv,bv,bv,bv,bv};
            const bf16* arow  = sAbf + l15 * KBF + half * 16;
            const bf16* wrowh = W2hi + (long)n * KBF + half * 16;
            const bf16* wrowl = W2lo + (long)n * KBF + half * 16;
            for (int k = 0; k < KBF; k += 32) {
                union { v16bf v; v8bf h[2]; } ua;
                ua.h[0] = *(const v8bf*)(arow + k);
                ua.h[1] = *(const v8bf*)(arow + k + 8);
                v16bf bh = *(const v16bf*)(wrowh + k);
                v16bf bl = *(const v16bf*)(wrowl + k);
                c = __builtin_amdgcn_wmma_f32_16x16x32_bf16(false, ua.v, false, bh,
                                                            (short)0, c, false, false);
                c = __builtin_amdgcn_wmma_f32_16x16x32_bf16(false, ua.v, false, bl,
                                                            (short)0, c, false, false);
            }
            float* mrow = &mem2[(nt * 16 + l15) * 16 + half * 8];
            v4f m0 = *(v4f*)(mrow);
            v4f m1 = *(v4f*)(mrow + 4);
            #pragma unroll
            for (int v = 0; v < 8; ++v) {
                float m  = (v < 4) ? m0[v & 3] : m1[v & 3];
                float mn = BETA * m + c[v] - ((m > 1.0f) ? 1.0f : 0.0f);
                if (v < 4) m0[v & 3] = mn; else m1[v & 3] = mn;
                sBbf[(v + half * 8) * KBF + n] = (bf16)((mn > 1.0f) ? 1.0f : 0.0f);
            }
            *(v4f*)(mrow) = m0;
            *(v4f*)(mrow + 4) = m1;
        }
        __syncthreads();

        // ---- Layer 3 (split-bf16 WMMA): cur = s2 @ W3^T + b3 ; LIF -> s3 ----
        if (wv < 2) {
            const int n = wv * 16 + l15;
            const float bv = b3p[n];
            v8f c = {bv,bv,bv,bv,bv,bv,bv,bv};
            const bf16* arow  = sBbf + l15 * KBF + half * 16;
            const bf16* wrowh = W3hi + (long)n * KBF + half * 16;
            const bf16* wrowl = W3lo + (long)n * KBF + half * 16;
            for (int k = 0; k < KBF; k += 32) {
                union { v16bf v; v8bf h[2]; } ua;
                ua.h[0] = *(const v8bf*)(arow + k);
                ua.h[1] = *(const v8bf*)(arow + k + 8);
                v16bf bh = *(const v16bf*)(wrowh + k);
                v16bf bl = *(const v16bf*)(wrowl + k);
                c = __builtin_amdgcn_wmma_f32_16x16x32_bf16(false, ua.v, false, bh,
                                                            (short)0, c, false, false);
                c = __builtin_amdgcn_wmma_f32_16x16x32_bf16(false, ua.v, false, bl,
                                                            (short)0, c, false, false);
            }
            #pragma unroll
            for (int v = 0; v < 8; ++v) {
                float m  = m3[v];
                float mn = BETA * m + c[v] - ((m > 1.0f) ? 1.0f : 0.0f);
                m3[v] = mn;
                s3[(v + half * 8) * HLP + n] = (mn > 1.0f) ? 1.0f : 0.0f;
            }
        }
        __syncthreads();

        // ---- Layer 4 (f32 WMMA): cur = s3 @ W4^T + b4 ; LIF -> output ----
        if (wv == 0) {
            const int n = l15;
            const float bv = b4p[n];
            v8f c = {bv,bv,bv,bv,bv,bv,bv,bv};
            const float* arow = &s3[l15 * HLP];
            for (int k = 0; k < HL; k += 4) {
                const int kk = k + half * 2;
                v2f a = *(const v2f*)(arow + kk);
                v2f b; b.x = W4T[(long)kk * OUTP + n]; b.y = W4T[(long)(kk + 1) * OUTP + n];
                c = __builtin_amdgcn_wmma_f32_16x16x4_f32(false, a, false, b,
                                                          (short)0, c, false, false);
            }
            #pragma unroll
            for (int v = 0; v < 8; ++v) {
                float m  = m4[v];
                float mn = BETA * m + c[v] - ((m > 1.0f) ? 1.0f : 0.0f);
                m4[v] = mn;
                float s = (mn > 1.0f) ? 1.0f : 0.0f;
                if (n < OUTN) {
                    const int row = v + half * 8;
                    long oi = (long)t * (1024L * OUTN) + (long)(r0 + row) * OUTN + n;
                    out[oi] = s;                                 // spk_rec
                    out[(long)NSTEPS * 1024L * OUTN + oi] = mn;  // mem_rec
                }
            }
        }
        __syncthreads();
    }
}

extern "C" void kernel_launch(void* const* d_in, const int* in_sizes, int n_in,
                              void* d_out, int out_size, void* d_ws, size_t ws_size,
                              hipStream_t stream) {
    const float* x  = (const float*)d_in[0];
    const float* W1 = (const float*)d_in[1];
    const float* b1 = (const float*)d_in[2];
    const float* W2 = (const float*)d_in[3];
    const float* b2 = (const float*)d_in[4];
    const float* W3 = (const float*)d_in[5];
    const float* b3 = (const float*)d_in[6];
    const float* W4 = (const float*)d_in[7];
    const float* b4 = (const float*)d_in[8];
    char*  ws  = (char*)d_ws;
    float* out = (float*)d_out;

    const int nbatch  = in_sizes[0] / (NSTEPS * NFREQ);  // 1024
    const int nblocks = nbatch / NROW;                   // 64 persistent WGs

    // 1) transpose / bf16-split / pad weights into workspace (L2-resident after)
    {
        int g = (PREP_MAX + 255) / 256;
        snn_prep<<<g, 256, 0, stream>>>(W1, b1, W2, b2, W3, b3, W4, b4, ws);
    }

    // 2) persistent recurrent kernel
    {
        size_t lds_bytes = (size_t)(2 * NTILES * 256) * 4  // mem1, mem2 (f32, tile-major)
                         + (size_t)(2 * NROW * KBF) * 2    // sAbf, sBbf (bf16)
                         + (size_t)(NROW * KFP) * 4        // xbuf
                         + (size_t)(NROW * HLP) * 4;       // s3
        snn_main<<<nblocks, 256, lds_bytes, stream>>>(x, ws, out);
    }
}
